// GraphTransformer_80272938762465
// MI455X (gfx1250) — compile-verified, matrix operand order checked by reading, not run on hardware
//
#include <hip/hip_runtime.h>

// ---------------------------------------------------------------------------
// Problem constants (from reference): NC=20, NU=16, K=128, C=2048, P=5, O=256
// ---------------------------------------------------------------------------
#define NC_ 20
#define NU_ 16
#define KK_ 128
#define CC_ 2048
#define PP_ 5
#define OO_ 256
#define GG_ (NC_ * NU_)   // 320 groups
#define MM_ (NU_ * KK_)   // 2048 rows per class in inter stage

typedef __bf16 bf16;
typedef __attribute__((ext_vector_type(16))) __bf16 v16bf;
typedef __attribute__((ext_vector_type(8)))  float  v8f;
typedef __attribute__((ext_vector_type(4)))  int    v4i_;

// Async global->LDS DMA path (gfx1250). Guarded: falls back to direct loads
// if this toolchain does not expose the builtins.
#if defined(__has_builtin)
#if __has_builtin(__builtin_amdgcn_global_load_async_to_lds_b128) && \
    __has_builtin(__builtin_amdgcn_s_wait_asynccnt)
#define HAVE_ASYNC_LDS 1
#endif
#endif
#ifndef HAVE_ASYNC_LDS
#define HAVE_ASYNC_LDS 0
#endif

// ---------------------------------------------------------------------------
// WMMA helpers (CDNA5 wave32, 16x16x32 bf16 -> f32)
// A fragment layout (ISA 7.12.2, 16-bit A 16x32):
//   lane = 16*h + m (h=0/1, m=row 0..15)
//   elems 0..7  : K = 8*h + 0..7      (contiguous 16 bytes)
//   elems 8..15 : K = 16 + 8*h + 0..7 (contiguous 16 bytes)
// B fragment symmetric with n = column. C/D: lane 16*h+n; vgpr i -> m = 8*h+i.
// ---------------------------------------------------------------------------
static __device__ __forceinline__ v16bf frag_rm(const bf16* __restrict__ rowp, int h) {
  union U { uint4 q[2]; v16bf v; } u;
  u.q[0] = *reinterpret_cast<const uint4*>(rowp + 8 * h);
  u.q[1] = *reinterpret_cast<const uint4*>(rowp + 16 + 8 * h);
  return u.v;
}

static __device__ __forceinline__ v16bf frag_cm(const bf16* __restrict__ colp, int ld, int h) {
  v16bf v;
#pragma unroll
  for (int i = 0; i < 8; ++i) {
    v[i]     = colp[(long long)(8 * h + i) * ld];
    v[i + 8] = colp[(long long)(16 + 8 * h + i) * ld];
  }
  return v;
}

static __device__ __forceinline__ v8f wmma_bf16(v16bf a, v16bf b, v8f c) {
  return __builtin_amdgcn_wmma_f32_16x16x32_bf16(false, a, false, b, (short)0, c,
                                                 false, false);
}

static __device__ __forceinline__ v8f vzero8() {
  v8f z;
#pragma unroll
  for (int i = 0; i < 8; ++i) z[i] = 0.0f;
  return z;
}

// ---------------------------------------------------------------------------
// Kernel 0: f32 -> bf16 bulk convert
// ---------------------------------------------------------------------------
__global__ void __launch_bounds__(256) k_cvt_bf16(const float* __restrict__ s,
                                                  bf16* __restrict__ d, long long n) {
  long long i = (long long)blockIdx.x * blockDim.x + threadIdx.x;
  if (i < n) d[i] = (bf16)s[i];
}

// ---------------------------------------------------------------------------
// Kernel 1: batched NT GEMM, C[b] = A[b] (MxKd) * B^T (NxKd), bf16 out.
// One wave32 per 32x64 output tile: 2 A-frags x 4 B-frags -> 8 WMMA / k-step.
// B shared across batch (weight matrix, resident in 192MB L2).
// ---------------------------------------------------------------------------
__global__ void __launch_bounds__(32) k_gemm_nt(const bf16* __restrict__ A,
                                                long long aStride,
                                                const bf16* __restrict__ B,
                                                bf16* __restrict__ Cb,
                                                long long cStride,
                                                int tilesN64, int Kdim) {
  const int lane = threadIdx.x;
  const int r = lane & 15, h = lane >> 4;
  const int tm = blockIdx.x / tilesN64, tn = blockIdx.x % tilesN64;
  const bf16* Arow = A + (long long)blockIdx.y * aStride + (long long)(tm * 32 + r) * Kdim;
  const bf16* Brow = B + (long long)(tn * 64 + r) * Kdim;

  v8f acc[2][4];
#pragma unroll
  for (int i = 0; i < 2; ++i)
#pragma unroll
    for (int j = 0; j < 4; ++j) acc[i][j] = vzero8();

  for (int k = 0; k < Kdim; k += 32) {
    v16bf a0 = frag_rm(Arow + k, h);
    v16bf a1 = frag_rm(Arow + (long long)16 * Kdim + k, h);
    v16bf b0 = frag_rm(Brow + k, h);
    v16bf b1 = frag_rm(Brow + (long long)16 * Kdim + k, h);
    v16bf b2 = frag_rm(Brow + (long long)32 * Kdim + k, h);
    v16bf b3 = frag_rm(Brow + (long long)48 * Kdim + k, h);
    acc[0][0] = wmma_bf16(a0, b0, acc[0][0]);
    acc[0][1] = wmma_bf16(a0, b1, acc[0][1]);
    acc[0][2] = wmma_bf16(a0, b2, acc[0][2]);
    acc[0][3] = wmma_bf16(a0, b3, acc[0][3]);
    acc[1][0] = wmma_bf16(a1, b0, acc[1][0]);
    acc[1][1] = wmma_bf16(a1, b1, acc[1][1]);
    acc[1][2] = wmma_bf16(a1, b2, acc[1][2]);
    acc[1][3] = wmma_bf16(a1, b3, acc[1][3]);
  }

  const int Ndim = tilesN64 * 64;
#pragma unroll
  for (int i = 0; i < 2; ++i)
#pragma unroll
    for (int j = 0; j < 4; ++j) {
      bf16* Cp = Cb + (long long)blockIdx.y * cStride +
                 (long long)(tm * 32 + i * 16) * Ndim + tn * 64 + j * 16 + r;
#pragma unroll
      for (int q = 0; q < 8; ++q) Cp[(long long)(q + 8 * h) * Ndim] = (bf16)acc[i][j][q];
    }
}

// ---------------------------------------------------------------------------
// Kernel 2: per-group inner attention scores + softmax.
// S = fe * fe^T (128x128, f32 in LDS via WMMA), att = softmax(S/16) -> bf16.
// One block (8 waves) per group; wave w computes rows [16w,16w+16).
// ---------------------------------------------------------------------------
__global__ void __launch_bounds__(256) k_attn_inner(const bf16* __restrict__ FE,
                                                    bf16* __restrict__ ATT) {
  __shared__ float S[128][129];
  const int tid = threadIdx.x;
  const int wave = tid >> 5, lane = tid & 31;
  const int r = lane & 15, h = lane >> 4;
  const long long g = blockIdx.x;
  const bf16* fe = FE + g * (long long)(KK_ * OO_);

  v8f acc[8];
#pragma unroll
  for (int t = 0; t < 8; ++t) acc[t] = vzero8();

  for (int ks = 0; ks < OO_; ks += 32) {
    v16bf a = frag_rm(fe + (long long)(wave * 16 + r) * OO_ + ks, h);
#pragma unroll
    for (int t = 0; t < 8; ++t) {
      v16bf b = frag_rm(fe + (long long)(t * 16 + r) * OO_ + ks, h);
      acc[t] = wmma_bf16(a, b, acc[t]);
    }
  }
#pragma unroll
  for (int t = 0; t < 8; ++t)
#pragma unroll
    for (int i = 0; i < 8; ++i) S[wave * 16 + i + 8 * h][t * 16 + r] = acc[t][i];
  __syncthreads();

  if (tid < 128) {
    const float scale = 0.0625f;  // 1/sqrt(256)
    float mx = -3.4e38f;
#pragma unroll 4
    for (int j = 0; j < 128; ++j) mx = fmaxf(mx, S[tid][j]);
    float sum = 0.0f;
#pragma unroll 4
    for (int j = 0; j < 128; ++j) {
      float e = __expf((S[tid][j] - mx) * scale);
      S[tid][j] = e;
      sum += e;
    }
    const float inv = 1.0f / sum;
    bf16* out = ATT + g * (long long)(KK_ * KK_) + (long long)tid * KK_;
#pragma unroll 4
    for (int j = 0; j < 128; ++j) out[j] = (bf16)(S[tid][j] * inv);
  }
}

// ---------------------------------------------------------------------------
// Kernel 3: agg = att (128x128) * X (128x2048), NN form (B is K-major).
// 32x64 tile per wave; B fragments gathered column-major.
// ---------------------------------------------------------------------------
__global__ void __launch_bounds__(32) k_gemm_agg(const bf16* __restrict__ ATT,
                                                 const bf16* __restrict__ X,
                                                 bf16* __restrict__ AGG) {
  const int lane = threadIdx.x;
  const int r = lane & 15, h = lane >> 4;
  const int tn = blockIdx.x & 31, tm = blockIdx.x >> 5;
  const long long g = blockIdx.y;
  const bf16* Arow = ATT + g * (long long)(KK_ * KK_) + (long long)(tm * 32 + r) * KK_;
  const bf16* Xg = X + g * (long long)(KK_ * CC_);

  v8f acc[2][4];
#pragma unroll
  for (int i = 0; i < 2; ++i)
#pragma unroll
    for (int j = 0; j < 4; ++j) acc[i][j] = vzero8();

  for (int k = 0; k < KK_; k += 32) {
    v16bf a0 = frag_rm(Arow + k, h);
    v16bf a1 = frag_rm(Arow + 16 * KK_ + k, h);
#pragma unroll
    for (int j = 0; j < 4; ++j) {
      v16bf b = frag_cm(Xg + (long long)k * CC_ + tn * 64 + j * 16 + r, CC_, h);
      acc[0][j] = wmma_bf16(a0, b, acc[0][j]);
      acc[1][j] = wmma_bf16(a1, b, acc[1][j]);
    }
  }
#pragma unroll
  for (int i = 0; i < 2; ++i)
#pragma unroll
    for (int j = 0; j < 4; ++j) {
      bf16* Cp = AGG + g * (long long)(KK_ * CC_) +
                 (long long)(tm * 32 + i * 16) * CC_ + tn * 64 + j * 16 + r;
#pragma unroll
      for (int q = 0; q < 8; ++q) Cp[(long long)(q + 8 * h) * CC_] = (bf16)acc[i][j][q];
    }
}

// ---------------------------------------------------------------------------
// Kernel 4: the big one (344 GFLOP). Y = relu(agg * Wt^T) + X  -> fa (bf16).
// 32x64 per wave (8 WMMA / 6 frag loads per k-step). Wt (8MB bf16) lives in L2.
// ---------------------------------------------------------------------------
__global__ void __launch_bounds__(32) k_gemm_trans(const bf16* __restrict__ AGG,
                                                   const bf16* __restrict__ Wt,
                                                   const float* __restrict__ Xf,
                                                   bf16* __restrict__ FA) {
  const int lane = threadIdx.x;
  const int r = lane & 15, h = lane >> 4;
  const int tn = blockIdx.x & 31, tm = blockIdx.x >> 5;
  const long long g = blockIdx.y;
  const bf16* Arow = AGG + g * (long long)(KK_ * CC_) + (long long)(tm * 32 + r) * CC_;
  const bf16* Brow = Wt + (long long)(tn * 64 + r) * CC_;

  v8f acc[2][4];
#pragma unroll
  for (int i = 0; i < 2; ++i)
#pragma unroll
    for (int j = 0; j < 4; ++j) acc[i][j] = vzero8();

  for (int k = 0; k < CC_; k += 32) {
    __builtin_prefetch(Arow + k + 128, 0, 1);
    v16bf a0 = frag_rm(Arow + k, h);
    v16bf a1 = frag_rm(Arow + 16 * CC_ + k, h);
    v16bf b0 = frag_rm(Brow + k, h);
    v16bf b1 = frag_rm(Brow + 16 * CC_ + k, h);
    v16bf b2 = frag_rm(Brow + 32 * CC_ + k, h);
    v16bf b3 = frag_rm(Brow + 48 * CC_ + k, h);
    acc[0][0] = wmma_bf16(a0, b0, acc[0][0]);
    acc[0][1] = wmma_bf16(a0, b1, acc[0][1]);
    acc[0][2] = wmma_bf16(a0, b2, acc[0][2]);
    acc[0][3] = wmma_bf16(a0, b3, acc[0][3]);
    acc[1][0] = wmma_bf16(a1, b0, acc[1][0]);
    acc[1][1] = wmma_bf16(a1, b1, acc[1][1]);
    acc[1][2] = wmma_bf16(a1, b2, acc[1][2]);
    acc[1][3] = wmma_bf16(a1, b3, acc[1][3]);
  }
#pragma unroll
  for (int i = 0; i < 2; ++i)
#pragma unroll
    for (int j = 0; j < 4; ++j)
#pragma unroll
      for (int q = 0; q < 8; ++q) {
        long long off = g * (long long)(KK_ * CC_) +
                        (long long)(tm * 32 + i * 16 + q + 8 * h) * CC_ +
                        tn * 64 + j * 16 + r;
        float y = fmaxf(acc[i][j][q], 0.0f) + Xf[off];  // relu + f32 residual
        FA[off] = (bf16)y;
      }
}

// ---------------------------------------------------------------------------
// Kernel 5: per-(class, prototype): pe = proto*Wi2^T (f32), logits = pe.fae,
// softmax over m=2048 -> ATT2 (f32). One block of 256 per (n,p).
// ---------------------------------------------------------------------------
__global__ void __launch_bounds__(256) k_logits(const float* __restrict__ protos,
                                                const float* __restrict__ Wi2,
                                                const bf16* __restrict__ FAE,
                                                float* __restrict__ ATT2) {
  __shared__ float prow[CC_];
  __shared__ float pe[OO_];
  __shared__ float lg[MM_];
  __shared__ float red[256];
  const int tid = threadIdx.x;
  const int n = blockIdx.x / PP_, p = blockIdx.x % PP_;

  const float* pr = protos + ((long long)n * PP_ + p) * CC_;
  for (int c = tid; c < CC_; c += 256) prow[c] = pr[c];
  __syncthreads();

  {
    const float* wrow = Wi2 + (long long)tid * CC_;
    float s = 0.0f;
#pragma unroll 8
    for (int c = 0; c < CC_; ++c) s += prow[c] * wrow[c];
    pe[tid] = s;
  }
  __syncthreads();

  const bf16* fae = FAE + (long long)n * MM_ * OO_;
  const float scale = 0.0625f;
  for (int m = tid; m < MM_; m += 256) {
    const bf16* frow = fae + (long long)m * OO_;
    float s = 0.0f;
#pragma unroll 8
    for (int o = 0; o < OO_; ++o) s += pe[o] * (float)frow[o];
    lg[m] = s * scale;
  }
  __syncthreads();

  float lmax = -3.4e38f;
  for (int m = tid; m < MM_; m += 256) lmax = fmaxf(lmax, lg[m]);
  red[tid] = lmax;
  __syncthreads();
  for (int s = 128; s > 0; s >>= 1) {
    if (tid < s) red[tid] = fmaxf(red[tid], red[tid + s]);
    __syncthreads();
  }
  const float mx = red[0];
  __syncthreads();
  float lsum = 0.0f;
  for (int m = tid; m < MM_; m += 256) {
    float e = __expf(lg[m] - mx);
    lg[m] = e;
    lsum += e;
  }
  red[tid] = lsum;
  __syncthreads();
  for (int s = 128; s > 0; s >>= 1) {
    if (tid < s) red[tid] += red[tid + s];
    __syncthreads();
  }
  const float inv = 1.0f / red[0];
  float* out = ATT2 + ((long long)n * PP_ + p) * MM_;
  for (int m = tid; m < MM_; m += 256) out[m] = lg[m] * inv;
}

// ---------------------------------------------------------------------------
// Kernel 6: un_protos[n,p,c] = sum_m att2[n,p,m] * fa[n,m,c]. Memory-bound.
// Double-buffered async global->LDS staging of 32-row fa slabs
// (global_load_async_to_lds_b128 + s_wait_asynccnt), overlapping DMA of
// slab k+1 with FMA consumption of slab k. Fallback: direct coalesced loads.
// ---------------------------------------------------------------------------
__global__ void __launch_bounds__(256) k_out(const float* __restrict__ ATT2,
                                             const bf16* __restrict__ FA,
                                             float* __restrict__ OUT) {
  __shared__ float a2[PP_][MM_];
  const int tid = threadIdx.x;
  const int n = blockIdx.y;
  const int cbase = blockIdx.x * 256;

  const float* src = ATT2 + (long long)n * PP_ * MM_;
  for (int idx = tid; idx < PP_ * MM_; idx += 256) a2[idx / MM_][idx % MM_] = src[idx];

  float acc[PP_];
#pragma unroll
  for (int p = 0; p < PP_; ++p) acc[p] = 0.0f;

#if HAVE_ASYNC_LDS
  __shared__ __align__(16) bf16 stage[2][32 * 256];  // 2 x 16KB slabs
  bf16* faBase = const_cast<bf16*>(FA) + (long long)n * MM_ * CC_ + cbase;

  auto issue = [&](int mb, int buf) {
#pragma unroll
    for (int j = 0; j < 4; ++j) {
      const int idx = tid + 256 * j;
      const int row = idx >> 5;         // 32 x b128 chunks per 512B row
      const int col = (idx & 31) << 3;  // bf16 elements
      bf16* g = faBase + (long long)(mb * 32 + row) * CC_ + col;
      bf16* l = &stage[buf][row * 256 + col];
      __builtin_amdgcn_global_load_async_to_lds_b128(
          (__attribute__((address_space(1))) v4i_*)g,
          (__attribute__((address_space(3))) v4i_*)l, 0, 0);
    }
  };

  issue(0, 0);
  for (int mb = 0; mb < MM_ / 32; ++mb) {
    const int cur = mb & 1;
    if (mb + 1 < MM_ / 32) {
      issue(mb + 1, cur ^ 1);
      __builtin_amdgcn_s_wait_asynccnt(4);  // this wave's cur-slab ops done
    } else {
      __builtin_amdgcn_s_wait_asynccnt(0);
    }
    __syncthreads();  // all waves' cur-slab data visible (and a2 on iter 0)
#pragma unroll 8
    for (int mm = 0; mm < 32; ++mm) {
      const float x = (float)stage[cur][mm * 256 + tid];
      const int m = mb * 32 + mm;
#pragma unroll
      for (int p = 0; p < PP_; ++p) acc[p] += a2[p][m] * x;
    }
    __syncthreads();  // reads done before this buffer is refilled
  }
#else
  __syncthreads();
  const bf16* fa = FA + (long long)n * MM_ * CC_ + cbase + tid;
  for (int m = 0; m < MM_; ++m) {
    const float x = (float)fa[(long long)m * CC_];
#pragma unroll
    for (int p = 0; p < PP_; ++p) acc[p] += a2[p][m] * x;
  }
#endif

#pragma unroll
  for (int p = 0; p < PP_; ++p)
    OUT[((long long)n * PP_ + p) * CC_ + cbase + tid] = acc[p];
}

// ---------------------------------------------------------------------------
// Host-side launcher
// ---------------------------------------------------------------------------
extern "C" void kernel_launch(void* const* d_in, const int* in_sizes, int n_in,
                              void* d_out, int out_size, void* d_ws, size_t ws_size,
                              hipStream_t stream) {
  (void)in_sizes; (void)n_in; (void)out_size; (void)ws_size;

  const float* topk  = (const float*)d_in[0];  // [20,16,128,2048]
  const float* prot  = (const float*)d_in[1];  // [20,5,2048]
  const float* w1    = (const float*)d_in[2];  // [256,2048]
  const float* wt    = (const float*)d_in[3];  // [2048,2048]
  const float* wi1   = (const float*)d_in[4];  // [256,2048]
  const float* wi2   = (const float*)d_in[5];  // [256,2048]
  float* out = (float*)d_out;                  // [20,5,2048]

  const long long nX   = (long long)GG_ * KK_ * CC_;  // 83,886,080
  const long long nW1  = (long long)OO_ * CC_;
  const long long nWt  = (long long)CC_ * CC_;
  const long long nWi1 = (long long)OO_ * CC_;
  const long long nFE  = (long long)GG_ * KK_ * OO_;
  const long long nATT = (long long)GG_ * KK_ * KK_;
  const long long nFAE = (long long)NC_ * MM_ * OO_;
  const long long nAT2 = (long long)NC_ * PP_ * MM_;

  char* p = (char*)d_ws;
  bf16* Xb   = (bf16*)p;  p += nX   * 2;
  bf16* W1b  = (bf16*)p;  p += nW1  * 2;
  bf16* Wtb  = (bf16*)p;  p += nWt  * 2;
  bf16* Wi1b = (bf16*)p;  p += nWi1 * 2;
  bf16* FEb  = (bf16*)p;  p += nFE  * 2;
  bf16* ATTb = (bf16*)p;  p += nATT * 2;
  bf16* AGGb = (bf16*)p;  p += nX   * 2;
  bf16* FAb  = (bf16*)p;  p += nX   * 2;
  bf16* FAEb = (bf16*)p;  p += nFAE * 2;
  float* AT2 = (float*)p; p += nAT2 * 4;

  // ---- Stage 0: bf16 conversions ----
  k_cvt_bf16<<<(unsigned)((nX   + 255) / 256), 256, 0, stream>>>(topk, Xb, nX);
  k_cvt_bf16<<<(unsigned)((nW1  + 255) / 256), 256, 0, stream>>>(w1,   W1b, nW1);
  k_cvt_bf16<<<(unsigned)((nWt  + 255) / 256), 256, 0, stream>>>(wt,   Wtb, nWt);
  k_cvt_bf16<<<(unsigned)((nWi1 + 255) / 256), 256, 0, stream>>>(wi1,  Wi1b, nWi1);

  // ---- Stage 1: fe = X * W1^T  (per-group [128,2048]x[2048,256]) ----
  k_gemm_nt<<<dim3((KK_ / 32) * (OO_ / 64), GG_), 32, 0, stream>>>(
      Xb, (long long)KK_ * CC_, W1b, FEb, (long long)KK_ * OO_, OO_ / 64, CC_);

  // ---- Stage 2: att = softmax(fe fe^T / 16) ----
  k_attn_inner<<<GG_, 256, 0, stream>>>(FEb, ATTb);

  // ---- Stage 3: agg = att * X ----
  k_gemm_agg<<<dim3((KK_ / 32) * (CC_ / 64), GG_), 32, 0, stream>>>(ATTb, Xb, AGGb);

  // ---- Stage 4: fa = relu(agg * Wt^T) + X ----
  k_gemm_trans<<<dim3((KK_ / 32) * (CC_ / 64), GG_), 32, 0, stream>>>(AGGb, Wtb, topk, FAb);

  // ---- Stage 5: fae = fa * Wi1^T (per-class [2048,2048]x[2048,256]) ----
  k_gemm_nt<<<dim3((MM_ / 32) * (OO_ / 64), NC_), 32, 0, stream>>>(
      FAb, (long long)MM_ * CC_, Wi1b, FAEb, (long long)MM_ * OO_, OO_ / 64, CC_);

  // ---- Stage 6: att2 = softmax(pe . fae / 16) ----
  k_logits<<<NC_ * PP_, 256, 0, stream>>>(prot, wi2, FAEb, AT2);

  // ---- Stage 7: un_protos = att2 * fa ----
  k_out<<<dim3(CC_ / 256, NC_), 256, 0, stream>>>(AT2, FAb, out);
}
